// scaled_dot_product_attention_18227841204461
// MI455X (gfx1250) — matrix operand order repaired and checked by reading, not compiled, not run on hardware
//
#include <hip/hip_runtime.h>

typedef __attribute__((ext_vector_type(16))) _Float16 v16h;
typedef __attribute__((ext_vector_type(8)))  _Float16 v8h;
typedef __attribute__((ext_vector_type(4)))  _Float16 v4h;
typedef __attribute__((ext_vector_type(8)))  float    v8f;

#define SEQ   1024
#define DIM   64
#define KTILE 64           // keys per iteration (amortize reduce + barriers)
#define QW    16           // query rows per wave
#define NWAVE 4
#define QBLK  (QW * NWAVE) // 64 query rows per block
#define NITER (SEQ / KTILE)

__device__ __forceinline__ float redmax16(float x) {
  x = fmaxf(x, __shfl_xor(x, 1, 32));
  x = fmaxf(x, __shfl_xor(x, 2, 32));
  x = fmaxf(x, __shfl_xor(x, 4, 32));
  x = fmaxf(x, __shfl_xor(x, 8, 32));
  return x;
}
__device__ __forceinline__ float redsum16(float x) {
  x += __shfl_xor(x, 1, 32);
  x += __shfl_xor(x, 2, 32);
  x += __shfl_xor(x, 4, 32);
  x += __shfl_xor(x, 8, 32);
  return x;
}

__global__ __launch_bounds__(128) void fa_fwd_wmma(const float* __restrict__ q,
                                                   const float* __restrict__ k,
                                                   const float* __restrict__ v,
                                                   float* __restrict__ out) {
  __shared__ __align__(32) _Float16 Ksh[KTILE * DIM];        // [key][d] f16
  __shared__ __align__(32) _Float16 Vtsh[DIM * KTILE];       // [d][key] f16 (V transposed)
  __shared__ __align__(32) _Float16 Psh[NWAVE][QW * KTILE];  // per-wave P scratch

  const int tid  = threadIdx.x;
  const int wave = tid >> 5;
  const int lane = tid & 31;
  const int n    = lane & 15;           // column index within 16 (B/C/D layouts)
  const int hiA  = (lane >> 4) * 8;     // half the A-frag K offset / C-frag row offset
  const int hiB  = (lane >> 4) * 16;    // B-frag K offset

  const int bh    = blockIdx.y;                 // head index (B*H)
  const int qrow0 = blockIdx.x * QBLK + wave * QW;

  const float scale = 0.125f;                   // 1/sqrt(64)

  const size_t head = (size_t)bh * SEQ * DIM;
  const float* qp = q + head;
  const float* kp = k + head;
  const float* vp = v + head;
  float*       op = out + head;

  // per-thread staging slot within the 64x64 tile (8 float4 chunks each for K and V)
  float4 kreg[8], vreg[8];
  auto loadKV = [&](int kt) {
    const int key0 = kt * KTILE;
    #pragma unroll
    for (int i = 0; i < 8; ++i) {
      const int idx = tid + i * 128;   // 1024 float4 chunks per 64x64 tile
      const int key = idx >> 4;        // 16 float4 per 64-dim row
      const int dc  = (idx & 15) * 4;
      kreg[i] = *reinterpret_cast<const float4*>(kp + (size_t)(key0 + key) * DIM + dc);
      vreg[i] = *reinterpret_cast<const float4*>(vp + (size_t)(key0 + key) * DIM + dc);
    }
  };
  auto storeKV = [&]() {
    #pragma unroll
    for (int i = 0; i < 8; ++i) {
      const int idx = tid + i * 128;
      const int key = idx >> 4;
      const int dc  = (idx & 15) * 4;
      v4h kh = { (_Float16)kreg[i].x, (_Float16)kreg[i].y,
                 (_Float16)kreg[i].z, (_Float16)kreg[i].w };
      *reinterpret_cast<v4h*>(&Ksh[key * DIM + dc]) = kh;
      Vtsh[(dc + 0) * KTILE + key] = (_Float16)vreg[i].x;
      Vtsh[(dc + 1) * KTILE + key] = (_Float16)vreg[i].y;
      Vtsh[(dc + 2) * KTILE + key] = (_Float16)vreg[i].z;
      Vtsh[(dc + 3) * KTILE + key] = (_Float16)vreg[i].w;
    }
  };

  // ---- Q A-fragments (pre-scaled, f16), two K-dim segments of 32 ----
  v16h qa[2];
  {
    const float* qr = qp + (size_t)(qrow0 + n) * DIM;
    #pragma unroll
    for (int s = 0; s < 2; ++s) {
      #pragma unroll
      for (int c = 0; c < 2; ++c) {
        const int d0 = 32 * s + 8 * c + 2 * hiA;
        float4 f0 = *reinterpret_cast<const float4*>(qr + d0);
        float4 f1 = *reinterpret_cast<const float4*>(qr + d0 + 4);
        qa[s][8 * c + 0] = (_Float16)(f0.x * scale);
        qa[s][8 * c + 1] = (_Float16)(f0.y * scale);
        qa[s][8 * c + 2] = (_Float16)(f0.z * scale);
        qa[s][8 * c + 3] = (_Float16)(f0.w * scale);
        qa[s][8 * c + 4] = (_Float16)(f1.x * scale);
        qa[s][8 * c + 5] = (_Float16)(f1.y * scale);
        qa[s][8 * c + 6] = (_Float16)(f1.z * scale);
        qa[s][8 * c + 7] = (_Float16)(f1.w * scale);
      }
    }
  }

  v8f o[4] = {{}, {}, {}, {}};          // output accumulators, d-tiles 0..3
  float rmax[8], rsum[8];
  #pragma unroll
  for (int r = 0; r < 8; ++r) { rmax[r] = -1e30f; rsum[r] = 0.0f; }

  loadKV(0);   // prologue: stage tile 0

  for (int kt = 0; kt < NITER; ++kt) {
    __syncthreads();                    // all waves done reading previous tile
    storeKV();                          // convert f32->f16, write K / V^T tiles
    if (kt + 1 < NITER) loadKV(kt + 1); // issue next tile's global loads early:
                                        // latency hidden behind barrier + compute
    __syncthreads();                    // LDS tile ready

    // ---- S = Q K^T : four 16x16 f32 tiles (keys 0-15,16-31,32-47,48-63) ----
    v8f sc[4];
    #pragma unroll
    for (int sub = 0; sub < 4; ++sub) {
      v16h b0 = *reinterpret_cast<const v16h*>(&Ksh[(16 * sub + n) * DIM +  0 + hiB]);
      v16h b1 = *reinterpret_cast<const v16h*>(&Ksh[(16 * sub + n) * DIM + 32 + hiB]);
      v8f s = {};
      s = __builtin_amdgcn_wmma_f32_16x16x32_f16(false, qa[0], false, b0, (short)0, s, false, false);
      s = __builtin_amdgcn_wmma_f32_16x16x32_f16(false, qa[1], false, b1, (short)0, s, false, false);
      sc[sub] = s;
    }

    // ---- online softmax (VGPR index = row, 16 lanes = keys; 4 keys/lane) ----
    _Float16* Pw = &Psh[wave][0];
    #pragma unroll
    for (int r = 0; r < 8; ++r) {
      float cm = fmaxf(fmaxf(sc[0][r], sc[1][r]), fmaxf(sc[2][r], sc[3][r]));
      float tm = redmax16(cm);
      float nm = fmaxf(rmax[r], tm);
      float corr = __expf(rmax[r] - nm);
      rmax[r] = nm;
      float p0 = __expf(sc[0][r] - nm);
      float p1 = __expf(sc[1][r] - nm);
      float p2 = __expf(sc[2][r] - nm);
      float p3 = __expf(sc[3][r] - nm);
      rsum[r] = rsum[r] * corr + ((p0 + p1) + (p2 + p3)); // lane-local partial
      o[0][r] *= corr; o[1][r] *= corr; o[2][r] *= corr; o[3][r] *= corr;
      Pw[(r + hiA) * KTILE +  0 + n] = (_Float16)p0;
      Pw[(r + hiA) * KTILE + 16 + n] = (_Float16)p1;
      Pw[(r + hiA) * KTILE + 32 + n] = (_Float16)p2;
      Pw[(r + hiA) * KTILE + 48 + n] = (_Float16)p3;
    }
    asm volatile("s_wait_dscnt 0" ::: "memory"); // CDNA5 split DS counter

    // ---- A-fragments of P (C layout -> A layout via per-wave LDS) ----
    v16h pa[2];
    #pragma unroll
    for (int j = 0; j < 2; ++j) {
      v8h lo = *reinterpret_cast<const v8h*>(&Pw[n * KTILE + 32 * j +     2 * hiA]);
      v8h hi = *reinterpret_cast<const v8h*>(&Pw[n * KTILE + 32 * j + 8 + 2 * hiA]);
      #pragma unroll
      for (int i = 0; i < 8; ++i) { pa[j][i] = lo[i]; pa[j][8 + i] = hi[i]; }
    }
    // ---- O += P V : 4 d-tiles x 2 key-segments ----
    #pragma unroll
    for (int t = 0; t < 4; ++t) {
      v16h vb0 = *reinterpret_cast<const v16h*>(&Vtsh[(16 * t + n) * KTILE +  0 + hiB]);
      v16h vb1 = *reinterpret_cast<const v16h*>(&Vtsh[(16 * t + n) * KTILE + 32 + hiB]);
      o[t] = __builtin_amdgcn_wmma_f32_16x16x32_f16(false, pa[0], false, vb0, (short)0, o[t], false, false);
      o[t] = __builtin_amdgcn_wmma_f32_16x16x32_f16(false, pa[1], false, vb1, (short)0, o[t], false, false);
    }
  }

  // ---- epilogue: reduce row sums, normalize, store ----
  #pragma unroll
  for (int r = 0; r < 8; ++r) {
    float s   = redsum16(rsum[r]);
    float inv = 1.0f / s;
    float* orow = op + (size_t)(qrow0 + r + hiA) * DIM;
    orow[ 0 + n] = o[0][r] * inv;
    orow[16 + n] = o[1][r] * inv;
    orow[32 + n] = o[2][r] * inv;
    orow[48 + n] = o[3][r] * inv;
  }
}

extern "C" void kernel_launch(void* const* d_in, const int* in_sizes, int n_in,
                              void* d_out, int out_size, void* d_ws, size_t ws_size,
                              hipStream_t stream) {
  const float* q = (const float*)d_in[0];
  const float* k = (const float*)d_in[1];
  const float* v = (const float*)d_in[2];
  float* out = (float*)d_out;
  (void)in_sizes; (void)n_in; (void)out_size; (void)d_ws; (void)ws_size;

  dim3 grid(SEQ / QBLK, 8 * 16); // 16 query tiles x (B*H)=128 heads
  fa_fwd_wmma<<<grid, 128, 0, stream>>>(q, k, v, out);
}
